// GlobalPointer_64544768525024
// MI455X (gfx1250) — matrix operand order repaired and checked by reading, not compiled, not run on hardware
//
#include <hip/hip_runtime.h>
#include <hip/hip_bf16.h>

typedef __bf16 bf16_t;
typedef __attribute__((ext_vector_type(16))) bf16_t v16bf;
typedef __attribute__((ext_vector_type(8)))  bf16_t v8bf;
typedef __attribute__((ext_vector_type(8)))  float  v8f;

#define NEGVAL (-1000000000000.0f)

// ---- fragment loaders (row-major bf16 source) -------------------------------
// A-matrix 16x32 bf16 (ISA 7.12.2): lane%16 = row M; lanes 0-15 hold K{0..7,16..23},
// lanes 16-31 hold K{8..15,24..31}. Two 16B chunks per lane.
__device__ __forceinline__ v16bf load_a_frag(const bf16_t* __restrict__ base, int ld,
                                             int row0, int k0, int lane) {
    const bf16_t* p = base + (size_t)(row0 + (lane & 15)) * ld + k0 + ((lane & 16) ? 8 : 0);
    union { v16bf v; v8bf h[2]; } u;
    u.h[0] = *(const v8bf*)(p);
    u.h[1] = *(const v8bf*)(p + 16);
    return u.v;
}
// B-matrix 32x16 bf16: lane%16 = col N; lanes 0-15 hold K{0..7,16..23},
// lanes 16-31 hold K{8..15,24..31} (same K order as A). Source stored as
// B^T row-major (row = N, contiguous K), so two 16B chunks per lane.
__device__ __forceinline__ v16bf load_b_frag(const bf16_t* __restrict__ base, int ld,
                                             int col0, int k0, int lane) {
    const bf16_t* p = base + (size_t)(col0 + (lane & 15)) * ld + k0 + ((lane & 16) ? 8 : 0);
    union { v16bf v; v8bf h[2]; } u;
    u.h[0] = *(const v8bf*)(p);
    u.h[1] = *(const v8bf*)(p + 16);
    return u.v;
}

#define WMMA_BF16(A, B, C) \
    __builtin_amdgcn_wmma_f32_16x16x32_bf16(false, (A), false, (B), (short)0, (C), false, false)

// ---- kernel 0a: h f32 -> bf16 ----------------------------------------------
__global__ void cvt_h_kernel(const float* __restrict__ src, bf16_t* __restrict__ dst, int n) {
    int i = blockIdx.x * blockDim.x + threadIdx.x;
    if (i < n) dst[i] = (bf16_t)src[i];
}

// ---- kernel 0b: W (768x1536) f32 -> W^T (1536x768) bf16 --------------------
__global__ void cvt_wt_kernel(const float* __restrict__ W, bf16_t* __restrict__ Wt) {
    int i = blockIdx.x * blockDim.x + threadIdx.x;   // over 1536*768
    if (i >= 1536 * 768) return;
    int n = i / 768, k = i - n * 768;
    Wt[i] = (bf16_t)W[(size_t)k * 1536 + n];
}

// ---- kernel 1: P = h @ W + b  (M=4096, N=1536, K=768), f32 out --------------
__global__ __launch_bounds__(256) void proj_gemm_kernel(const bf16_t* __restrict__ hb,
                                                        const bf16_t* __restrict__ wt,
                                                        const float* __restrict__ bias,
                                                        float* __restrict__ P) {
    const int lane = threadIdx.x & 31;
    const int wid  = threadIdx.x >> 5;
    const int wm = wid & 3, wn = wid >> 2;
    const int m0 = blockIdx.y * 128 + wm * 32;   // 4 waves along M
    const int n0 = blockIdx.x * 128 + wn * 64;   // 2 waves along N

    v8f acc[2][4] = {};
    for (int k0 = 0; k0 < 768; k0 += 32) {
        v16bf a0 = load_a_frag(hb, 768, m0,      k0, lane);
        v16bf a1 = load_a_frag(hb, 768, m0 + 16, k0, lane);
        v16bf b0 = load_b_frag(wt, 768, n0,      k0, lane);
        v16bf b1 = load_b_frag(wt, 768, n0 + 16, k0, lane);
        v16bf b2 = load_b_frag(wt, 768, n0 + 32, k0, lane);
        v16bf b3 = load_b_frag(wt, 768, n0 + 48, k0, lane);
        acc[0][0] = WMMA_BF16(a0, b0, acc[0][0]);
        acc[0][1] = WMMA_BF16(a0, b1, acc[0][1]);
        acc[0][2] = WMMA_BF16(a0, b2, acc[0][2]);
        acc[0][3] = WMMA_BF16(a0, b3, acc[0][3]);
        acc[1][0] = WMMA_BF16(a1, b0, acc[1][0]);
        acc[1][1] = WMMA_BF16(a1, b1, acc[1][1]);
        acc[1][2] = WMMA_BF16(a1, b2, acc[1][2]);
        acc[1][3] = WMMA_BF16(a1, b3, acc[1][3]);
    }
    // C/D layout: lane%16 = N, row M = g + 8*(lane>=16)
    const int rbase = m0 + ((lane & 16) ? 8 : 0);
    const int cb    = n0 + (lane & 15);
    #pragma unroll
    for (int fn = 0; fn < 4; ++fn) {
        const int c = cb + fn * 16;
        const float bv = bias[c];
        #pragma unroll
        for (int fm = 0; fm < 2; ++fm) {
            #pragma unroll
            for (int g = 0; g < 8; ++g) {
                int r = rbase + fm * 16 + g;
                P[(size_t)r * 1536 + c] = acc[fm][fn][g] + bv;
            }
        }
    }
}

// ---- kernel 2: RoPE + pack to q/k bf16 (B,H,L,64) ---------------------------
__global__ void rope_pack_kernel(const float* __restrict__ P,
                                 bf16_t* __restrict__ q, bf16_t* __restrict__ k) {
    int idx = blockIdx.x * blockDim.x + threadIdx.x;   // over 4096*768 pairs
    if (idx >= 4096 * 768) return;
    int m   = idx / 768;         // flattened (b,l)
    int p   = idx - m * 768;     // pair id within row
    int head = p >> 6;
    int rem  = p & 63;
    int qk   = rem >> 5;
    int i    = rem & 31;
    int l    = m & 1023;
    int bidx = m >> 10;
    const float* src = P + (size_t)m * 1536 + head * 128 + qk * 64 + 2 * i;
    float x1 = src[0], x2 = src[1];
    // inv_freq = 10000^(-i/32) ; ln(10000)/32 = 0.287823136624...
    float ang = (float)l * __expf(-0.28782313662425572f * (float)i);
    float s, c;
    __sincosf(ang, &s, &c);
    float r1 = x1 * c - x2 * s;
    float r2 = x1 * s + x2 * c;
    bf16_t* dst = (qk ? k : q) + (((size_t)(bidx * 12 + head)) * 1024 + l) * 64 + 2 * i;
    dst[0] = (bf16_t)r1;
    dst[1] = (bf16_t)r2;
}

// ---- kernel 3: logits = q @ k^T * 0.125, masked, NT-stored ------------------
__global__ __launch_bounds__(256) void attn_logits_kernel(const bf16_t* __restrict__ q,
                                                          const bf16_t* __restrict__ kmat,
                                                          const int* __restrict__ tmask,
                                                          float* __restrict__ out) {
    const int lane = threadIdx.x & 31;
    const int wid  = threadIdx.x >> 5;
    const int wm = wid & 3, wn = wid >> 2;
    const int bh   = blockIdx.z;          // b*12 + head
    const int bidx = bh / 12;
    const bf16_t* qb = q    + (size_t)bh * 1024 * 64;
    const bf16_t* kb = kmat + (size_t)bh * 1024 * 64;
    float* ob = out + (size_t)bh * 1024 * 1024;
    const int* mk = tmask + bidx * 1024;
    const int m0 = blockIdx.y * 128 + wm * 32;
    const int n0 = blockIdx.x * 128 + wn * 64;

    v8f acc[2][4] = {};
    #pragma unroll
    for (int k0 = 0; k0 < 64; k0 += 32) {
        v16bf a0 = load_a_frag(qb, 64, m0,      k0, lane);
        v16bf a1 = load_a_frag(qb, 64, m0 + 16, k0, lane);
        v16bf b0 = load_b_frag(kb, 64, n0,      k0, lane);
        v16bf b1 = load_b_frag(kb, 64, n0 + 16, k0, lane);
        v16bf b2 = load_b_frag(kb, 64, n0 + 32, k0, lane);
        v16bf b3 = load_b_frag(kb, 64, n0 + 48, k0, lane);
        acc[0][0] = WMMA_BF16(a0, b0, acc[0][0]);
        acc[0][1] = WMMA_BF16(a0, b1, acc[0][1]);
        acc[0][2] = WMMA_BF16(a0, b2, acc[0][2]);
        acc[0][3] = WMMA_BF16(a0, b3, acc[0][3]);
        acc[1][0] = WMMA_BF16(a1, b0, acc[1][0]);
        acc[1][1] = WMMA_BF16(a1, b1, acc[1][1]);
        acc[1][2] = WMMA_BF16(a1, b2, acc[1][2]);
        acc[1][3] = WMMA_BF16(a1, b3, acc[1][3]);
    }

    const int rbase = m0 + ((lane & 16) ? 8 : 0);
    const int cb    = n0 + (lane & 15);
    int cm[4];
    #pragma unroll
    for (int fn = 0; fn < 4; ++fn) cm[fn] = mk[cb + fn * 16];

    #pragma unroll
    for (int fm = 0; fm < 2; ++fm) {
        #pragma unroll
        for (int g = 0; g < 8; ++g) {
            const int r  = rbase + fm * 16 + g;
            const int rm = mk[r];
            float* rowp = ob + (size_t)r * 1024;
            #pragma unroll
            for (int fn = 0; fn < 4; ++fn) {
                const int c = cb + fn * 16;
                float v = acc[fm][fn][g] * 0.125f;
                bool keep = (rm != 0) && (cm[fn] != 0) && (c >= r);
                __builtin_nontemporal_store(keep ? v : NEGVAL, rowp + c);
            }
        }
    }
}

extern "C" void kernel_launch(void* const* d_in, const int* in_sizes, int n_in,
                              void* d_out, int out_size, void* d_ws, size_t ws_size,
                              hipStream_t stream) {
    const float* h    = (const float*)d_in[0];   // (4,1024,768)
    const float* W    = (const float*)d_in[1];   // (768,1536)
    const float* bias = (const float*)d_in[2];   // (1536,)
    const int*   tm   = (const int*)d_in[3];     // (4,1024)
    float* out = (float*)d_out;                  // (4,12,1024,1024)

    char* ws = (char*)d_ws;
    const size_t off_hb = 0;                                 // 4096*768*2  = 6,291,456
    const size_t off_wt = off_hb + (size_t)4096 * 768 * 2;   // 1536*768*2  = 2,359,296
    const size_t off_P  = off_wt + (size_t)1536 * 768 * 2;   // 4096*1536*4 = 25,165,824
    const size_t off_q  = off_P  + (size_t)4096 * 1536 * 4;  // 48*1024*64*2
    const size_t off_k  = off_q  + (size_t)48 * 1024 * 64 * 2;
    bf16_t* hb = (bf16_t*)(ws + off_hb);
    bf16_t* wt = (bf16_t*)(ws + off_wt);
    float*  P  = (float*)(ws + off_P);
    bf16_t* qw = (bf16_t*)(ws + off_q);
    bf16_t* kw = (bf16_t*)(ws + off_k);

    const int NH = 4096 * 768;
    cvt_h_kernel<<<(NH + 255) / 256, 256, 0, stream>>>(h, hb, NH);
    cvt_wt_kernel<<<(1536 * 768 + 255) / 256, 256, 0, stream>>>(W, wt);
    proj_gemm_kernel<<<dim3(12, 32), 256, 0, stream>>>(hb, wt, bias, P);
    rope_pack_kernel<<<(4096 * 768 + 255) / 256, 256, 0, stream>>>(P, qw, kw);
    attn_logits_kernel<<<dim3(8, 8, 48), 256, 0, stream>>>(qw, kw, tm, out);
}
